// HyenaOnly_11304353923766
// MI455X (gfx1250) — compile-verified, matrix operand order checked by reading, not run on hardware
//
#include <hip/hip_runtime.h>
#include <hip/hip_bf16.h>
#include <math.h>
#include <stdint.h>

// ---------------- problem constants ----------------
constexpr int Bb   = 8;
constexpr int Ll   = 8192;
constexpr int Dd   = 256;
constexpr int FILT = 127;
constexpr int EMBc = 33;
constexpr int HIDc = 64;
constexpr int FFT_N   = 8192;   // == Ll
constexpr int FFT_LOG = 13;
constexpr int FFT_T   = 256;

typedef __attribute__((ext_vector_type(16))) __bf16 v16bf;
typedef __attribute__((ext_vector_type(8)))  __bf16 v8bf;
typedef __attribute__((ext_vector_type(4)))  __bf16 v4bf;
typedef __attribute__((ext_vector_type(8)))  float  v8f;
typedef __attribute__((ext_vector_type(4)))  float  v4f;

// CDNA5 16-bit fragment: lane's 16 elements = two contiguous 8-element K-runs:
// K = hi*8 + [0..7]  and  K = 16 + hi*8 + [0..7]   (hi = lane>>4)
__device__ __forceinline__ v16bf load_frag(const __bf16* rowbase, int hi) {
  v8bf lo8 = *(const v8bf*)(rowbase + hi * 8);
  v8bf hi8 = *(const v8bf*)(rowbase + 16 + hi * 8);
  return __builtin_shufflevector(lo8, hi8, 0, 1, 2, 3, 4, 5, 6, 7,
                                 8, 9, 10, 11, 12, 13, 14, 15);
}

// ---------------- f32[K,N] -> bf16 transposed [N,K] ----------------
__global__ void k_cvt_bf16_T(const float* __restrict__ in, __bf16* __restrict__ out,
                             int K, int N) {
  int i = blockIdx.x * blockDim.x + threadIdx.x;
  if (i >= K * N) return;
  int k = i / N, n = i % N;
  out[(size_t)n * K + k] = (__bf16)in[i];
}

// ---------------- bf16 WMMA GEMM: C[M,N] = A_f32[M,K] * W^T_bf16[N,K] + bias -------
// 256 threads = 8 waves (2x4); block tile 64x128; each wave computes 32x32.
// K-step 64 (two WMMA sub-steps per stage). B tile staged via async global->LDS DMA.
__global__ __launch_bounds__(256)
void k_gemm_wmma(const float* __restrict__ A, const __bf16* __restrict__ BwT,
                 const float* __restrict__ bias, float* __restrict__ C,
                 int M, int N, int K) {
  __shared__ __attribute__((aligned(16))) __bf16 As[64][72];    // rows x K (padded)
  __shared__ __attribute__((aligned(16))) __bf16 Bst[128][72];  // cols x K (padded)
  const int tid  = threadIdx.x;
  const int lane = tid & 31, wave = tid >> 5;
  const int wm = wave >> 2, wn = wave & 3;          // 2 x 4 waves
  const int m0 = blockIdx.y * 64, n0 = blockIdx.x * 128;
  const int lo = lane & 15, hi = lane >> 4;

  v8f acc[2][2];
#pragma unroll
  for (int mi = 0; mi < 2; ++mi)
#pragma unroll
    for (int ni = 0; ni < 2; ++ni)
      acc[mi][ni] = (v8f){0.f, 0.f, 0.f, 0.f, 0.f, 0.f, 0.f, 0.f};

  for (int k0 = 0; k0 < K; k0 += 64) {
    // --- stage B tile: 128 cols x 64 k bf16, async global->LDS 16B copies ---
#pragma unroll
    for (int t = 0; t < 4; ++t) {
      int i = tid + t * 256;
      int col = i >> 3, kc = (i & 7) * 8;
      const __bf16* gp = &BwT[(size_t)(n0 + col) * K + (k0 + kc)];
      unsigned lds = (unsigned)(uintptr_t)&Bst[col][kc];
      asm volatile("global_load_async_to_lds_b128 %0, %1, off"
                   :: "v"(lds), "v"((unsigned long long)(uintptr_t)gp)
                   : "memory");
    }
    // --- stage A tile: 64 rows x 64 k, batched float4 loads then bf16 stores ---
    v4f av[4];
#pragma unroll
    for (int t = 0; t < 4; ++t) {
      int i = tid + t * 256;
      int r = i >> 4, c4 = (i & 15) * 4;
      av[t] = *(const v4f*)&A[(size_t)(m0 + r) * K + (k0 + c4)];
    }
#pragma unroll
    for (int t = 0; t < 4; ++t) {
      int i = tid + t * 256;
      int r = i >> 4, c4 = (i & 15) * 4;
      v4bf ab;
      ab[0] = (__bf16)av[t][0]; ab[1] = (__bf16)av[t][1];
      ab[2] = (__bf16)av[t][2]; ab[3] = (__bf16)av[t][3];
      *(v4bf*)&As[r][c4] = ab;
    }
    if (k0 + 64 < K)
      __builtin_prefetch(&A[(size_t)(m0 + (tid >> 2)) * K + k0 + 64], 0, 1);
    asm volatile("s_wait_asynccnt 0x0" ::: "memory");
    __syncthreads();

#pragma unroll
    for (int ks = 0; ks < 64; ks += 32) {
      v16bf a0 = load_frag(&As[wm * 32 + lo][ks], hi);
      v16bf a1 = load_frag(&As[wm * 32 + 16 + lo][ks], hi);
      v16bf b0 = load_frag(&Bst[wn * 32 + lo][ks], hi);
      v16bf b1 = load_frag(&Bst[wn * 32 + 16 + lo][ks], hi);

      acc[0][0] = __builtin_amdgcn_wmma_f32_16x16x32_bf16(false, a0, false, b0,
                                                          (short)0, acc[0][0], false, false);
      acc[0][1] = __builtin_amdgcn_wmma_f32_16x16x32_bf16(false, a0, false, b1,
                                                          (short)0, acc[0][1], false, false);
      acc[1][0] = __builtin_amdgcn_wmma_f32_16x16x32_bf16(false, a1, false, b0,
                                                          (short)0, acc[1][0], false, false);
      acc[1][1] = __builtin_amdgcn_wmma_f32_16x16x32_bf16(false, a1, false, b1,
                                                          (short)0, acc[1][1], false, false);
    }
    __syncthreads();
  }

  // C/D layout: VGPR r -> row = r + hi*8, col = lane&15
#pragma unroll
  for (int mi = 0; mi < 2; ++mi)
#pragma unroll
    for (int ni = 0; ni < 2; ++ni) {
      int colg = n0 + wn * 32 + ni * 16 + lo;
      float bv = bias[colg];
#pragma unroll
      for (int r = 0; r < 8; ++r) {
        int rowg = m0 + wm * 32 + mi * 16 + hi * 8 + r;
        C[(size_t)rowg * N + colg] = acc[mi][ni][r] + bv;
      }
    }
}

// ---------------- dwconv3 (SAME) + split + kv = k*v ----------------
__global__ void k_dwconv_qkv(const float* __restrict__ z, const float* __restrict__ scw,
                             const float* __restrict__ scb,
                             float* __restrict__ q, float* __restrict__ kv) {
  size_t idx = (size_t)blockIdx.x * blockDim.x + threadIdx.x;
  size_t total = (size_t)Bb * Ll * Dd;
  if (idx >= total) return;
  int d = (int)(idx % Dd);
  int l = (int)((idx / Dd) % Ll);
  size_t bl = idx / Dd;                     // b*L + l
  float r[3];
#pragma unroll
  for (int p = 0; p < 3; ++p) {
    int c = p * Dd + d;
    size_t base = bl * (size_t)(3 * Dd) + c;
    float vm = (l > 0)      ? z[base - 3 * Dd] : 0.f;
    float v0 = z[base];
    float vp = (l < Ll - 1) ? z[base + 3 * Dd] : 0.f;
    r[p] = scw[c * 3 + 0] * vm + scw[c * 3 + 1] * v0 + scw[c * 3 + 2] * vp + scb[c];
  }
  q[idx]  = r[0];
  kv[idx] = r[1] * r[2];
}

// ---------------- explicit 127-tap filter with decay ----------------
__global__ void k_mr_filter(const float* __restrict__ filt, const float* __restrict__ decay,
                            float* __restrict__ h) {
  int i = blockIdx.x * blockDim.x + threadIdx.x;
  if (i >= FILT * Dd) return;
  int f = i / Dd, d = i % Dd;
  h[i] = filt[i] * __expf(-((float)f / (float)(FILT - 1)) * fabsf(decay[d]));
}

// ---------------- causal FIR (block 1 long conv) + per-channel skip ----------------
__global__ __launch_bounds__(256)
void k_fir127(const float* __restrict__ kv, const float* __restrict__ h,
              const float* __restrict__ cb, float* __restrict__ y) {
  int bt = blockIdx.x;            // b*L + t
  int t  = bt % Ll;
  int d  = threadIdx.x;
  size_t rowbase = (size_t)bt * Dd;
  float acc = kv[rowbase + d] * cb[d];
  int taps = (t + 1 < FILT) ? (t + 1) : FILT;
  for (int tau = 0; tau < taps; ++tau)
    acc += h[tau * Dd + d] * kv[(size_t)(bt - tau) * Dd + d];
  y[rowbase + d] = acc;
}

// ---------------- implicit filter: pos-features -> Sin-MLP -> decay ----------------
__global__ __launch_bounds__(256)
void k_li_filter(const float* __restrict__ w1, const float* __restrict__ b1,
                 const float* __restrict__ w2, const float* __restrict__ decay,
                 float* __restrict__ h) {
  const int t   = blockIdx.x;
  const int tid = threadIdx.x;
  __shared__ float zf[EMBc];
  __shared__ float hid[HIDc];
  const float wt = 2.f * 3.14159265358979323846f * (float)t / (float)Ll;
  if (tid == 0) {
    zf[0] = (float)t / (float)Ll;
  } else if (tid <= 16) {
    int k = tid - 1;
    float f = 1e-4f + (float)k * (15.f - 1e-4f) / 15.f;
    zf[tid] = __cosf(f * wt);
  } else if (tid <= 32) {
    int k = tid - 17;
    float f = 1e-4f + (float)k * (15.f - 1e-4f) / 15.f;
    zf[tid] = __sinf(f * wt);
  }
  __syncthreads();
  if (tid < HIDc) {
    float acc = b1[tid];
    for (int e = 0; e < EMBc; ++e) acc += zf[e] * w1[e * HIDc + tid];
    hid[tid] = __sinf(10.f * acc);
  }
  __syncthreads();
  float acc = 0.f;
  for (int e = 0; e < HIDc; ++e) acc += hid[e] * w2[e * Dd + tid];
  float tm = (float)t / (float)(Ll - 1);
  h[(size_t)t * Dd + tid] = acc * __expf(-tm * fabsf(decay[tid]));
}

// ---------------- in-LDS radix-2 FFT, N=8192, 256 threads ----------------
__device__ void fft_lds(float* re, float* im, float sign) {
  const int tid = threadIdx.x;
  for (int i = tid; i < FFT_N; i += FFT_T) {
    int j = (int)(__brev((unsigned)i) >> (32 - FFT_LOG));
    if (j > i) {
      float tr = re[i]; re[i] = re[j]; re[j] = tr;
      float ti = im[i]; im[i] = im[j]; im[j] = ti;
    }
  }
  __syncthreads();
  for (int s = 0; s < FFT_LOG; ++s) {
    const int half = 1 << s;
    for (int j = tid; j < FFT_N / 2; j += FFT_T) {
      const int k  = j & (half - 1);
      const int i0 = ((j >> s) << (s + 1)) + k;
      const int i1 = i0 + half;
      const float ang = sign * (-6.283185307179586f) * (float)k / (float)(2 * half);
      float sn, c;
      __sincosf(ang, &sn, &c);
      const float xr = re[i1], xi = im[i1];
      const float tr = xr * c - xi * sn;
      const float ti = xr * sn + xi * c;
      const float ur = re[i0], ui = im[i0];
      re[i0] = ur + tr; im[i0] = ui + ti;
      re[i1] = ur - tr; im[i1] = ui - ti;
    }
    __syncthreads();
  }
}

// filter spectrum: h[L,D] column d -> HF[d][0..N)
__global__ void k_fft_filter(const float* __restrict__ h, float2* __restrict__ HF) {
  extern __shared__ float smem[];
  float* re = smem;
  float* im = smem + FFT_N;
  const int d = blockIdx.x;
  for (int i = threadIdx.x; i < FFT_N; i += FFT_T) {
    re[i] = h[(size_t)i * Dd + d];
    im[i] = 0.f;
  }
  __syncthreads();
  fft_lds(re, im, 1.f);
  for (int i = threadIdx.x; i < FFT_N; i += FFT_T)
    HF[(size_t)d * FFT_N + i] = make_float2(re[i], im[i]);
}

// circular conv per (b,d) row + per-channel skip bias
__global__ void k_fftconv(const float* __restrict__ kvT, const float2* __restrict__ HF,
                          const float* __restrict__ cb, float* __restrict__ yT) {
  extern __shared__ float smem[];
  float* re = smem;
  float* im = smem + FFT_N;
  const int bd = blockIdx.x;       // b*D + d
  const int d  = bd % Dd;
  const float* src = kvT + (size_t)bd * FFT_N;
  for (int i = threadIdx.x; i < FFT_N; i += FFT_T) { re[i] = src[i]; im[i] = 0.f; }
  __syncthreads();
  fft_lds(re, im, 1.f);
  for (int i = threadIdx.x; i < FFT_N; i += FFT_T) {
    float2 Hf = HF[(size_t)d * FFT_N + i];
    float xr = re[i], xi = im[i];
    re[i] = xr * Hf.x - xi * Hf.y;
    im[i] = xr * Hf.y + xi * Hf.x;
  }
  __syncthreads();
  fft_lds(re, im, -1.f);
  const float inv = 1.f / (float)FFT_N;
  const float cbd = cb[d];
  for (int i = threadIdx.x; i < FFT_N; i += FFT_T)
    yT[(size_t)bd * FFT_N + i] = re[i] * inv + src[i] * cbd;
}

// ---------------- batched 32x32 tiled transpose: in[b][R][C] -> out[b][C][R] --------
__global__ __launch_bounds__(256)
void k_transpose(const float* __restrict__ in, float* __restrict__ out, int R, int C) {
  __shared__ float tile[32][33];
  const int b  = blockIdx.z;
  const int c0 = blockIdx.x * 32;
  const int r0 = blockIdx.y * 32;
  const size_t base = (size_t)b * R * C;
  for (int i = threadIdx.y; i < 32; i += 8)
    tile[i][threadIdx.x] = in[base + (size_t)(r0 + i) * C + (c0 + threadIdx.x)];
  __syncthreads();
  for (int i = threadIdx.y; i < 32; i += 8)
    out[base + (size_t)(c0 + i) * R + (r0 + threadIdx.x)] = tile[threadIdx.x][i];
}

// ---------------- gate: u = q * y ----------------
__global__ void k_gate(const float* __restrict__ q, const float* __restrict__ y,
                       float* __restrict__ u, int n) {
  int i = blockIdx.x * blockDim.x + threadIdx.x;
  if (i < n) u[i] = q[i] * y[i];
}

// ---------------- RMSNorm over D=256, one row per block ----------------
__global__ __launch_bounds__(256)
void k_rmsnorm(const float* __restrict__ x, const float* __restrict__ w,
               float* __restrict__ out) {
  const int row = blockIdx.x;
  const int tid = threadIdx.x;
  const size_t base = (size_t)row * Dd;
  float v = x[base + tid];
  float s = v * v;
#pragma unroll
  for (int o = 16; o > 0; o >>= 1) s += __shfl_xor(s, o, 32);
  __shared__ float part[8];
  __shared__ float scale;
  if ((tid & 31) == 0) part[tid >> 5] = s;
  __syncthreads();
  if (tid == 0) {
    float t = 0.f;
    for (int i = 0; i < 8; ++i) t += part[i];
    scale = rsqrtf(t / (float)Dd + 1e-8f);
  }
  __syncthreads();
  out[base + tid] = v * scale * w[tid];
}

// ---------------- host orchestration ----------------
extern "C" void kernel_launch(void* const* d_in, const int* in_sizes, int n_in,
                              void* d_out, int out_size, void* d_ws, size_t ws_size,
                              hipStream_t stream) {
  (void)in_sizes; (void)n_in; (void)out_size; (void)ws_size;
  const float* x      = (const float*)d_in[0];
  const float* norm_w = (const float*)d_in[32];

  const int IW[3]  = {1, 10, 21}, IB[3]  = {2, 11, 22};
  const int SW[3]  = {3, 12, 23}, SB[3]  = {4, 13, 24};
  const int DEC[3] = {6, 17, 28}, CBi[3] = {7, 18, 29};
  const int OW[3]  = {8, 19, 30}, OB[3]  = {9, 20, 31};
  const int W1[3]  = {0, 14, 25}, B1[3]  = {0, 15, 26}, W2[3] = {0, 16, 27};

  const size_t BL  = (size_t)Bb * Ll;
  const size_t BLD = BL * Dd;

  // workspace layout (z region reused for kvT / yT / u after dwconv consumes z)
  float*  z   = (float*)d_ws;          // BL * 3D
  float*  kvT = z;                     // BLD   (reuse)
  float*  yT  = z + BLD;               // BLD   (reuse)
  float*  u   = z + 2 * BLD;           // BLD   (reuse)
  float*  q   = z + 3 * BLD;
  float*  kv  = q + BLD;
  float*  y   = kv + BLD;
  float*  xb0 = y + BLD;
  float*  xb1 = xb0 + BLD;
  float*  h   = xb1 + BLD;                               // Ll * Dd
  float2* HF  = (float2*)(h + (size_t)Ll * Dd);          // Dd * Ll complex
  __bf16* wAT = (__bf16*)(HF + (size_t)Dd * FFT_N);      // [3Dd][Dd] transposed
  __bf16* wOT = wAT + (size_t)3 * Dd * Dd;               // [Dd][Dd] transposed

  const float* xin = x;
  float* outs[3] = {xb0, xb1, xb0};
  const size_t shbytes = 2 * (size_t)FFT_N * sizeof(float);

  for (int blk = 0; blk < 3; ++blk) {
    const float* in_w  = (const float*)d_in[IW[blk]];
    const float* in_b  = (const float*)d_in[IB[blk]];
    const float* sc_w  = (const float*)d_in[SW[blk]];
    const float* sc_b  = (const float*)d_in[SB[blk]];
    const float* decay = (const float*)d_in[DEC[blk]];
    const float* cb    = (const float*)d_in[CBi[blk]];
    const float* out_w = (const float*)d_in[OW[blk]];
    const float* out_b = (const float*)d_in[OB[blk]];

    const int Nin = 3 * Dd;

    // weights -> transposed bf16 [N][K]
    k_cvt_bf16_T<<<(Dd * Nin + 255) / 256, 256, 0, stream>>>(in_w, wAT, Dd, Nin);
    k_cvt_bf16_T<<<(Dd * Dd + 255) / 256, 256, 0, stream>>>(out_w, wOT, Dd, Dd);

    // in-projection GEMM: [BL, D] x [D, 3D]
    k_gemm_wmma<<<dim3(Nin / 128, (unsigned)(BL / 64)), 256, 0, stream>>>(
        xin, wAT, in_b, z, (int)BL, Nin, Dd);

    // short conv + split + kv
    k_dwconv_qkv<<<(unsigned)((BLD + 255) / 256), 256, 0, stream>>>(z, sc_w, sc_b, q, kv);

    if (blk == 0) {
      const float* filt = (const float*)d_in[5];
      k_mr_filter<<<(FILT * Dd + 255) / 256, 256, 0, stream>>>(filt, decay, h);
      k_fir127<<<(unsigned)BL, 256, 0, stream>>>(kv, h, cb, y);
    } else {
      const float* w1 = (const float*)d_in[W1[blk]];
      const float* b1 = (const float*)d_in[B1[blk]];
      const float* w2 = (const float*)d_in[W2[blk]];
      k_li_filter<<<Ll, 256, 0, stream>>>(w1, b1, w2, decay, h);
      k_fft_filter<<<Dd, 256, shbytes, stream>>>(h, HF);
      k_transpose<<<dim3(Dd / 32, Ll / 32, Bb), dim3(32, 8), 0, stream>>>(kv, kvT, Ll, Dd);
      k_fftconv<<<Bb * Dd, 256, shbytes, stream>>>(kvT, HF, cb, yT);
      k_transpose<<<dim3(Ll / 32, Dd / 32, Bb), dim3(32, 8), 0, stream>>>(yT, y, Dd, Ll);
    }

    // gate then out-projection GEMM: [BL, D] x [D, D]
    k_gate<<<(unsigned)((BLD + 255) / 256), 256, 0, stream>>>(q, y, u, (int)BLD);
    k_gemm_wmma<<<dim3(Dd / 128, (unsigned)(BL / 64)), 256, 0, stream>>>(
        u, wOT, out_b, outs[blk], (int)BL, Dd, Dd);

    xin = outs[blk];
  }

  k_rmsnorm<<<(unsigned)BL, 256, 0, stream>>>(xin, norm_w, (float*)d_out);
}
